// MultiQueryAttention_38001870635858
// MI455X (gfx1250) — matrix operand order repaired and checked by reading, not compiled, not run on hardware
//
#include <hip/hip_runtime.h>
#include <hip/hip_bf16.h>
#include <math.h>

#define HIDDEN 2048
#define NHEADS 16
#define HDIM   128
#define BATCH  2
#define SEQ    2048

typedef __attribute__((ext_vector_type(16))) __bf16       v16bf;
typedef __attribute__((ext_vector_type(8)))  float        v8f;
typedef __attribute__((ext_vector_type(4)))  unsigned int v4u;
typedef __attribute__((ext_vector_type(8)))  int          v8i;
typedef __attribute__((ext_vector_type(4)))  int          v4i;

union ABu { v16bf v; uint4 q[2]; };

#define WMMA_BF16(A_, B_, C_) \
  __builtin_amdgcn_wmma_f32_16x16x32_bf16(false, (A_), false, (B_), (short)0, (C_), false, false)

// A fragment (16x32, MxK bf16): lane<16 -> row m0+lane, K = {k0..k0+7, k0+16..k0+23}
//                               lane>=16 -> same rows, K shifted by +8.
__device__ __forceinline__ v16bf ldA(const __bf16* rowptr, int k0, int half8) {
  ABu u;
  u.q[0] = *(const uint4*)(rowptr + k0 + half8);
  u.q[1] = *(const uint4*)(rowptr + k0 + half8 + 16);
  return u.v;
}

// B fragment (32x16, KxN bf16): lane holds column n0+(lane&15), K = {k0..k0+7,
// k0+16..k0+23}, shifted by +8 for lane>=16.  rowptr = W + (n0+(lane&15))*ldw.
__device__ __forceinline__ v16bf ldB(const __bf16* rowptr, int k0, int half16) {
  ABu u;
  u.q[0] = *(const uint4*)(rowptr + k0 + (half16 >> 1));
  u.q[1] = *(const uint4*)(rowptr + k0 + (half16 >> 1) + 16);
  return u.v;
}

// XOR-swizzle reduction helper (group-of-32 mode, and_mask=0x1f, xor=MASK).
template<int MASK>
__device__ __forceinline__ float swz(float v) {
  return __int_as_float(__builtin_amdgcn_ds_swizzle(__float_as_int(v), 0x1F | (MASK << 10)));
}

// ------------------------------------------------------------- TDM issue ----
// 2-D tile load via the Tensor Data Mover. All operands are block-uniform.
// D# group0: count=1 | lds_addr | global_addr[56:0] | type=2.
// D# group1: data_size=2B, tensor_dim0/1, tile_dim0/1, tensor_dim0_stride.
__device__ __forceinline__ void tdm_load_2d(const void* gptr, unsigned lds_off,
                                            unsigned dim0, unsigned dim1,
                                            unsigned tile0, unsigned tile1,
                                            unsigned stride0) {
  unsigned long long ga = (unsigned long long)(size_t)gptr;
  v4u g0;
  g0[0] = 1u;                                                  // count = 1
  g0[1] = lds_off;                                             // LDS byte addr
  g0[2] = (unsigned)ga;                                        // gaddr[31:0]
  g0[3] = (unsigned)((ga >> 32) & 0x01FFFFFFu) | 0x80000000u;  // gaddr[56:32] | type=2
  v8i g1;
  g1[0] = (int)(1u << 16);                                     // data_size = 2 bytes
  g1[1] = (int)((dim0 & 0xFFFFu) << 16);                       // tensor_dim0 lo
  g1[2] = (int)(((dim0 >> 16) & 0xFFFFu) | ((dim1 & 0xFFFFu) << 16));
  g1[3] = (int)(((dim1 >> 16) & 0xFFFFu) | (tile0 << 16));     // tile_dim0
  g1[4] = (int)(tile1 & 0xFFFFu);                              // tile_dim1, tile_dim2=0
  g1[5] = (int)stride0;                                        // dim0 stride lo
  g1[6] = 0;                                                   // stride hi / dim1 stride
  g1[7] = 0;
  v4i z = {0, 0, 0, 0};
#if defined(__clang_major__) && (__clang_major__ >= 23)
  v8i z8 = {0, 0, 0, 0, 0, 0, 0, 0};
  __builtin_amdgcn_tensor_load_to_lds(g0, g1, z, z, z8, 0);
#else
  __builtin_amdgcn_tensor_load_to_lds(g0, g1, z, z, 0);
#endif
}

// ---------------------------------------------------------------- casts ----
__global__ void cast_f32_bf16_kernel(const float* __restrict__ x,
                                     __bf16* __restrict__ y, int n) {
  for (int i = blockIdx.x * blockDim.x + threadIdx.x; i < n;
       i += gridDim.x * blockDim.x)
    y[i] = (__bf16)x[i];
}

// v [B,S,D] f32  ->  vt [B,D,S] bf16
__global__ void transpose_cast_v_kernel(const float* __restrict__ v,
                                        __bf16* __restrict__ vt) {
  const int total = BATCH * SEQ * HDIM;
  for (int i = blockIdx.x * blockDim.x + threadIdx.x; i < total;
       i += gridDim.x * blockDim.x) {
    int d = i % HDIM;
    int s = (i / HDIM) % SEQ;
    int b = i / (HDIM * SEQ);
    vt[((size_t)b * HDIM + d) * SEQ + s] = (__bf16)v[i];
  }
}

// broadcast single KV head to [B,H,S,D] fp32 outputs
__global__ void kv_broadcast_kernel(const float* __restrict__ kf,
                                    const float* __restrict__ vf,
                                    float* __restrict__ keyo,
                                    float* __restrict__ valo) {
  const int total = BATCH * NHEADS * SEQ * HDIM;
  for (int i = blockIdx.x * blockDim.x + threadIdx.x; i < total;
       i += gridDim.x * blockDim.x) {
    int d = i % HDIM;
    int s = (i / HDIM) % SEQ;
    int b = i / (HDIM * SEQ * NHEADS);
    size_t src = ((size_t)b * SEQ + s) * HDIM + d;
    keyo[i] = kf[src];
    valo[i] = vf[src];
  }
}

// ------------------------------------------------ GEMM: C = A[M,K] * W[N,K]^T
// One wave -> 16x64 output tile (4 x v8f accumulators).
template<bool WRITE_F32, bool WRITE_BF16>
__global__ __launch_bounds__(256)
void gemm_bf16_wmma(const __bf16* __restrict__ A, const __bf16* __restrict__ W,
                    float* __restrict__ Cf, __bf16* __restrict__ Cb,
                    int M, int N, int K) {
  const int lane = threadIdx.x & 31;
  const int wave = blockIdx.x * (blockDim.x >> 5) + (threadIdx.x >> 5);
  const int ntiles = N >> 6;
  const int mt = wave / ntiles;
  const int nt = wave % ntiles;
  if (mt * 16 >= M) return;                       // wave-uniform
  const int m0 = mt << 4;
  const int n0 = nt << 6;
  const int lidx   = lane & 15;
  const int half8  = (lane >> 4) << 3;
  const int half16 = (lane >> 4) << 4;

  const __bf16* arow = A + (size_t)(m0 + lidx) * K;
  const __bf16* w0   = W + (size_t)(n0 +  0 + lidx) * K;
  const __bf16* w1   = W + (size_t)(n0 + 16 + lidx) * K;
  const __bf16* w2   = W + (size_t)(n0 + 32 + lidx) * K;
  const __bf16* w3   = W + (size_t)(n0 + 48 + lidx) * K;

  v8f acc0 = {}, acc1 = {}, acc2 = {}, acc3 = {};
  for (int k0 = 0; k0 < K; k0 += 32) {
    if (k0 + 64 < K) {                            // CDNA5 global_prefetch_b8
      __builtin_prefetch(arow + k0 + 64, 0, 1);
      __builtin_prefetch(w0 + k0 + 64, 0, 1);
    }
    v16bf a = ldA(arow, k0, half8);
    acc0 = WMMA_BF16(a, ldB(w0, k0, half16), acc0);
    acc1 = WMMA_BF16(a, ldB(w1, k0, half16), acc1);
    acc2 = WMMA_BF16(a, ldB(w2, k0, half16), acc2);
    acc3 = WMMA_BF16(a, ldB(w3, k0, half16), acc3);
  }

  const int row  = m0 + half8;
  const int col0 = n0 + lidx;
#pragma unroll
  for (int r = 0; r < 8; ++r) {
    size_t off = (size_t)(row + r) * N;
    if (WRITE_F32) {
      Cf[off + col0 +  0] = acc0[r];
      Cf[off + col0 + 16] = acc1[r];
      Cf[off + col0 + 32] = acc2[r];
      Cf[off + col0 + 48] = acc3[r];
    }
    if (WRITE_BF16) {
      Cb[off + col0 +  0] = (__bf16)acc0[r];
      Cb[off + col0 + 16] = (__bf16)acc1[r];
      Cb[off + col0 + 32] = (__bf16)acc2[r];
      Cb[off + col0 + 48] = (__bf16)acc3[r];
    }
  }
}

// ------------------------------------------- flash-style MQA attention ----
// Block = 4 waves, all same (b,h), 4 consecutive 16-row M tiles (64 rows).
// K and V^T tiles for each 32-key step are staged into LDS once per block by
// the Tensor Data Mover (double-buffered, TENSORcnt + barrier synchronized),
// then consumed by all 4 waves.
__global__ __launch_bounds__(128)
void mqa_attention_kernel(const __bf16* __restrict__ qb,   // [B,S,HIDDEN]
                          const __bf16* __restrict__ kb,   // [B,S,HDIM]
                          const __bf16* __restrict__ vtb,  // [B,HDIM,S]
                          const float*  __restrict__ mask, // [B,S]
                          __bf16* __restrict__ attnb) {    // [B,S,HIDDEN]
  __shared__ __align__(128) __bf16 ktile[2][32 * HDIM];    // 2 x 8 KB
  __shared__ __align__(128) __bf16 vtile[2][HDIM * 32];    // 2 x 8 KB
  __shared__ __align__(32)  __bf16 plds_all[4][16 * 32];   // 4 x 1 KB

  const int lane  = threadIdx.x & 31;
  const int wslot = threadIdx.x >> 5;
  __bf16* plds = plds_all[wslot];

  const int mblk = blockIdx.x & (SEQ / 64 - 1);   // 32 M blocks of 64 rows
  const int bh   = blockIdx.x >> 5;
  const int h    = bh & (NHEADS - 1);
  const int b    = bh >> 4;

  const int lidx   = lane & 15;
  const int half8  = (lane >> 4) << 3;
  const int half16 = (lane >> 4) << 4;
  const int m0 = mblk * 64 + wslot * 16;

  const __bf16* qrow  = qb  + ((size_t)b * SEQ + m0 + lidx) * HIDDEN + h * HDIM;
  const __bf16* kbase = kb  + (size_t)b * SEQ * HDIM;
  const __bf16* vbase = vtb + (size_t)b * HDIM * SEQ;
  const float*  mrowg = mask + (size_t)b * SEQ;

  v16bf qa[4];
#pragma unroll
  for (int t = 0; t < 4; ++t) qa[t] = ldA(qrow, t * 32, half8);

  v8f o[8] = {};
  float rm[8], rl[8];
#pragma unroll
  for (int r = 0; r < 8; ++r) { rm[r] = -__builtin_inff(); rl[r] = 0.f; }

  const float scale = 0.08838834764831845f;          // 1/sqrt(128)
  const float NEG   = -3.4028234663852886e38f;       // finfo(f32).min

  // Preload buffer 0: K rows [0,32) and V^T cols [0,32).
  if (wslot == 0) {
    tdm_load_2d(kbase, (unsigned)(size_t)&ktile[0][0],
                HDIM, SEQ, HDIM, 32, HDIM);
    tdm_load_2d(vbase, (unsigned)(size_t)&vtile[0][0],
                SEQ, HDIM, 32, HDIM, SEQ);
  }

  for (int j0 = 0; j0 < SEQ; j0 += 32) {
    const int buf = (j0 >> 5) & 1;
    if (wslot == 0) __builtin_amdgcn_s_wait_tensorcnt(0);
    __syncthreads();                              // buffer `buf` ready for all
    if (wslot == 0 && (j0 + 32) < SEQ) {          // prefetch next step
      tdm_load_2d(kbase + (size_t)(j0 + 32) * HDIM,
                  (unsigned)(size_t)&ktile[buf ^ 1][0],
                  HDIM, SEQ, HDIM, 32, HDIM);
      tdm_load_2d(vbase + (size_t)(j0 + 32),
                  (unsigned)(size_t)&vtile[buf ^ 1][0],
                  SEQ, HDIM, 32, HDIM, SEQ);
    }

    const __bf16* kb0 = &ktile[buf][0] + (size_t)lidx * HDIM;
    const __bf16* kb1 = &ktile[buf][0] + (size_t)(16 + lidx) * HDIM;
    v8f s0 = {}, s1 = {};
#pragma unroll
    for (int t = 0; t < 4; ++t) {
      s0 = WMMA_BF16(qa[t], ldB(kb0, t * 32, half16), s0);
      s1 = WMMA_BF16(qa[t], ldB(kb1, t * 32, half16), s1);
    }

    const float madd0 = (1.0f - mrowg[j0 +      lidx]) * NEG;
    const float madd1 = (1.0f - mrowg[j0 + 16 + lidx]) * NEG;

    float af[8];
#pragma unroll
    for (int r = 0; r < 8; ++r) {
      float a = s0[r] * scale + madd0;
      float c = s1[r] * scale + madd1;
      float mx = fmaxf(a, c);
      mx = fmaxf(mx, swz<1>(mx)); mx = fmaxf(mx, swz<2>(mx));
      mx = fmaxf(mx, swz<4>(mx)); mx = fmaxf(mx, swz<8>(mx));
      float nm = fmaxf(rm[r], mx);
      float alpha = __expf(rm[r] - nm);
      float p0 = __expf(a - nm);
      float p1 = __expf(c - nm);
      s0[r] = p0; s1[r] = p1;
      float sm = p0 + p1;
      sm += swz<1>(sm); sm += swz<2>(sm); sm += swz<4>(sm); sm += swz<8>(sm);
      rl[r] = rl[r] * alpha + sm;
      rm[r] = nm;
      af[r] = alpha;
    }

#pragma unroll
    for (int t = 0; t < 8; ++t)
#pragma unroll
      for (int r = 0; r < 8; ++r) o[t][r] *= af[r];

    // C-layout probs -> LDS -> A-layout bf16 fragment (wave-private region)
#pragma unroll
    for (int r = 0; r < 8; ++r) {
      int row = r + half8;
      plds[row * 32 +      lidx] = (__bf16)s0[r];
      plds[row * 32 + 16 + lidx] = (__bf16)s1[r];
    }
    asm volatile("s_wait_dscnt 0x0" ::: "memory");
    v16bf pa = ldA(plds + (size_t)lidx * 32, 0, half8);

    const __bf16* vb = &vtile[buf][0];
#pragma unroll
    for (int t = 0; t < 8; ++t)
      o[t] = WMMA_BF16(pa, ldB(vb + (size_t)(t * 16 + lidx) * 32, 0, half16), o[t]);
  }

#pragma unroll
  for (int r = 0; r < 8; ++r) rl[r] = 1.0f / rl[r];
  const size_t orow = (size_t)b * SEQ + m0 + half8;
#pragma unroll
  for (int t = 0; t < 8; ++t)
#pragma unroll
    for (int r = 0; r < 8; ++r)
      attnb[(orow + r) * HIDDEN + h * HDIM + t * 16 + lidx] =
          (__bf16)(o[t][r] * rl[r]);
}

// --------------------------------------------------------------- launch ----
extern "C" void kernel_launch(void* const* d_in, const int* in_sizes, int n_in,
                              void* d_out, int out_size, void* d_ws, size_t ws_size,
                              hipStream_t stream) {
  const float* hs   = (const float*)d_in[0];
  const float* mask = (const float*)d_in[1];
  const float* Wq   = (const float*)d_in[2];
  const float* Wk   = (const float*)d_in[3];
  const float* Wv   = (const float*)d_in[4];
  const float* Wo   = (const float*)d_in[5];

  float* out  = (float*)d_out;
  float* keyo = out  + (size_t)BATCH * SEQ * HIDDEN;
  float* valo = keyo + (size_t)BATCH * NHEADS * SEQ * HDIM;

  char* ws = (char*)d_ws;
  auto alloc = [&](size_t bytes) {
    char* p = ws;
    ws += (bytes + 255) & ~(size_t)255;
    return p;
  };
  __bf16* hsb   = (__bf16*)alloc((size_t)BATCH * SEQ * HIDDEN * 2);
  __bf16* wqb   = (__bf16*)alloc((size_t)HIDDEN * HIDDEN * 2);
  __bf16* wkb   = (__bf16*)alloc((size_t)HDIM * HIDDEN * 2);
  __bf16* wvb   = (__bf16*)alloc((size_t)HDIM * HIDDEN * 2);
  __bf16* wob   = (__bf16*)alloc((size_t)HIDDEN * HIDDEN * 2);
  __bf16* qbuf  = (__bf16*)alloc((size_t)BATCH * SEQ * HIDDEN * 2);
  float*  kf    = (float*) alloc((size_t)BATCH * SEQ * HDIM * 4);
  float*  vf    = (float*) alloc((size_t)BATCH * SEQ * HDIM * 4);
  __bf16* kbuf  = (__bf16*)alloc((size_t)BATCH * SEQ * HDIM * 2);
  __bf16* vtb   = (__bf16*)alloc((size_t)BATCH * HDIM * SEQ * 2);
  __bf16* attnb = (__bf16*)alloc((size_t)BATCH * SEQ * HIDDEN * 2);

  cast_f32_bf16_kernel<<<1024, 256, 0, stream>>>(hs, hsb, BATCH * SEQ * HIDDEN);
  cast_f32_bf16_kernel<<<512,  256, 0, stream>>>(Wq, wqb, HIDDEN * HIDDEN);
  cast_f32_bf16_kernel<<<64,   256, 0, stream>>>(Wk, wkb, HDIM * HIDDEN);
  cast_f32_bf16_kernel<<<64,   256, 0, stream>>>(Wv, wvb, HDIM * HIDDEN);
  cast_f32_bf16_kernel<<<512,  256, 0, stream>>>(Wo, wob, HIDDEN * HIDDEN);

  const int Mq = BATCH * SEQ;  // 4096

  // Q projection -> bf16 [B*S, HIDDEN]
  gemm_bf16_wmma<false, true><<<(Mq / 16) * (HIDDEN / 64) / 8, 256, 0, stream>>>(
      hsb, wqb, nullptr, qbuf, Mq, HIDDEN, HIDDEN);
  // K projection -> f32 (for KV-cache output) + bf16 (for attention)
  gemm_bf16_wmma<true, true><<<(Mq / 16) * (HDIM / 64) / 8, 256, 0, stream>>>(
      hsb, wkb, kf, kbuf, Mq, HDIM, HIDDEN);
  // V projection -> f32
  gemm_bf16_wmma<true, false><<<(Mq / 16) * (HDIM / 64) / 8, 256, 0, stream>>>(
      hsb, wvb, vf, nullptr, Mq, HDIM, HIDDEN);

  transpose_cast_v_kernel<<<512, 256, 0, stream>>>(vf, vtb);
  kv_broadcast_kernel<<<2048, 256, 0, stream>>>(kf, vf, keyo, valo);

  mqa_attention_kernel<<<1024, 128, 0, stream>>>(qbuf, kbuf, vtb, mask, attnb);

  // Output projection -> fp32 directly into d_out
  gemm_bf16_wmma<true, false><<<(Mq / 16) * (HIDDEN / 64) / 8, 256, 0, stream>>>(
      attnb, wob, out, nullptr, Mq, HIDDEN, HIDDEN);
}
